// SparseMoE_24515673326231
// MI455X (gfx1250) — matrix operand order repaired and checked by reading, not compiled, not run on hardware
//
#include <hip/hip_runtime.h>
#include <hip/hip_bf16.h>

typedef __attribute__((ext_vector_type(16))) _Float16 v16h;
typedef __attribute__((ext_vector_type(8)))  _Float16 v8h;
typedef __attribute__((ext_vector_type(8)))  float    v8f;

#define B_  16
#define S_  1024
#define D_  512
#define H_  2048
#define E_  16
#define K_  4
#define T_  (B_ * S_)

// ---------------------------------------------------------------------------
// Helpers
// ---------------------------------------------------------------------------
__device__ __forceinline__ v16h load_a_frag(const _Float16* p) {
    // 16-bit A-fragment: lane holds row m, K in {ak0..ak0+15} (contiguous)
    v8h lo = *(const v8h*)p;
    v8h hi = *(const v8h*)(p + 8);
    return __builtin_shufflevector(lo, hi,
        0,1,2,3,4,5,6,7,8,9,10,11,12,13,14,15);
}

__device__ __forceinline__ v8f wmma_f16(v16h a, v16h b, v8f c) {
    return __builtin_amdgcn_wmma_f32_16x16x32_f16(
        false, a, false, b, (short)0, c, false, false);
}

// ---------------------------------------------------------------------------
// Router: logits = x @ router_w + router_b ; top-4 ; softmax ; build per-expert
// token lists (counts/ids/gates) in workspace via atomics.
// ---------------------------------------------------------------------------
__global__ __launch_bounds__(256) void moe_router_kernel(
    const float* __restrict__ x,
    const float* __restrict__ rw,
    const float* __restrict__ rb,
    int*   __restrict__ counts,
    int*   __restrict__ lists,
    float* __restrict__ gates)
{
    __shared__ float xs[16][D_];   // 32 KB
    __shared__ float lg[16][E_];

    const int tid  = threadIdx.x;
    const int tok0 = blockIdx.x * 16;

    for (int i = tid; i < 16 * D_; i += 256) {
        int m = i >> 9, k = i & (D_ - 1);
        xs[m][k] = x[(size_t)(tok0 + m) * D_ + k];
    }
    __syncthreads();

    const int m = tid >> 4;
    const int e = tid & 15;
    float acc = rb[e];
    for (int k = 0; k < D_; ++k) acc += xs[m][k] * rw[k * E_ + e];
    lg[m][e] = acc;
    __syncthreads();

    if (e == 0) {
        float v[E_];
        #pragma unroll
        for (int i = 0; i < E_; ++i) v[i] = lg[m][i];
        int   idx[K_];
        float val[K_];
        #pragma unroll
        for (int kk = 0; kk < K_; ++kk) {
            int bi = 0; float bv = -3.4e38f;
            #pragma unroll
            for (int i = 0; i < E_; ++i)
                if (v[i] > bv) { bv = v[i]; bi = i; }
            idx[kk] = bi; val[kk] = bv; v[bi] = -3.4e38f;
        }
        const float mx = val[0];
        float p[K_], s = 0.f;
        #pragma unroll
        for (int kk = 0; kk < K_; ++kk) { p[kk] = __expf(val[kk] - mx); s += p[kk]; }
        const float inv = 1.f / s;
        const int tok = tok0 + m;
        #pragma unroll
        for (int kk = 0; kk < K_; ++kk) {
            int ex  = idx[kk];
            int pos = atomicAdd(&counts[ex], 1);
            lists[ex * T_ + pos] = tok;
            gates[ex * T_ + pos] = p[kk] * inv;
        }
    }
}

// ---------------------------------------------------------------------------
// x (f32, [T,D]) -> xh (f16, [T,D]); 8 elements per thread.
// ---------------------------------------------------------------------------
__global__ __launch_bounds__(256) void cvt_x_kernel(
    const float* __restrict__ x, _Float16* __restrict__ xh)
{
    const size_t g = (size_t)blockIdx.x * 256 + threadIdx.x;
    const size_t i = g * 8;
    v8h o;
    #pragma unroll
    for (int j = 0; j < 8; ++j) o[j] = (_Float16)x[i + j];
    *(v8h*)(xh + i) = o;
}

// ---------------------------------------------------------------------------
// Pack f32 weight [E][KROWS][NCOLS] into WMMA B-fragment order f16:
//   dst[frag][lane][i],  frag = (e*(NCOLS/16) + nchunk)*(KROWS/32) + ktile
//   lane: n = nchunk*16 + lane%16 ; k = ktile*32 + 16*(lane/16) + i
// One thread emits one lane-slice (16 f16 = 32 B contiguous).
// ---------------------------------------------------------------------------
template<int NCOLS, int KROWS>
__global__ __launch_bounds__(256) void pack_w_kernel(
    const float* __restrict__ src, _Float16* __restrict__ dst)
{
    const size_t g    = (size_t)blockIdx.x * 256 + threadIdx.x;
    const int    lane = (int)(g & 31);
    const size_t frag = g >> 5;
    const size_t nfrag = (size_t)E_ * (NCOLS / 16) * (KROWS / 32);
    if (frag >= nfrag) return;

    const int ktile = (int)(frag % (KROWS / 32));
    size_t tmp      = frag / (KROWS / 32);
    const int nch   = (int)(tmp % (NCOLS / 16));
    const int e     = (int)(tmp / (NCOLS / 16));

    const int nlo = lane & 15;
    const int bk0 = (lane >> 4) * 16;
    const float* sp = src + (size_t)e * KROWS * NCOLS
                          + (size_t)(ktile * 32 + bk0) * NCOLS
                          + nch * 16 + nlo;
    v16h o;
    #pragma unroll
    for (int i = 0; i < 16; ++i) o[i] = (_Float16)sp[(size_t)i * NCOLS];
    *(v16h*)(dst + frag * 512 + (size_t)lane * 16) = o;
}

// ---------------------------------------------------------------------------
// Fast expert FFN: one workgroup per (expert, 32-token tile).
// A fragments straight from global f16 xh (per-lane row pointers).
// B fragments are single 32B contiguous loads from packed weights.
// H processed in four 512-column slabs; h slab lives in 32 KB LDS.
// 256 threads = 8 waves (wave32).
// ---------------------------------------------------------------------------
__global__ __launch_bounds__(256) void moe_ffn_fast_kernel(
    const _Float16* __restrict__ xh,
    const _Float16* __restrict__ w1p,
    const float*    __restrict__ b1,
    const _Float16* __restrict__ w2p,
    const float*    __restrict__ b2,
    const int*      __restrict__ counts,
    const int*      __restrict__ lists,
    const float*    __restrict__ gates,
    float*          __restrict__ out)
{
    __shared__ __align__(16) _Float16 hs[32 * 512];   // 32 KB h slab
    __shared__ int   toks[32];
    __shared__ float gts[32];

    const int e    = blockIdx.x >> 9;        // T/32 = 512 tiles per expert
    const int tile = blockIdx.x & 511;
    const int cnt  = counts[e];
    const int base = tile * 32;
    if (base >= cnt) return;                 // block-uniform exit

    const int tid = threadIdx.x;
    if (tid < 32) {
        int  i     = base + tid;
        bool valid = (i < cnt);
        int  j     = valid ? i : base;       // pad rows: first token, gate 0
        toks[tid] = lists[e * T_ + j];
        gts[tid]  = valid ? gates[e * T_ + i] : 0.f;
    }
    __syncthreads();

    const int lane   = tid & 31;
    const int wave   = tid >> 5;
    const int nlo    = lane & 15;
    const int hihalf = lane >> 4;
    const int ak0    = hihalf * 16;          // A-frag K offset for this lane

    const float* __restrict__ b1e = b1 + e * H_;
    const float* __restrict__ b2e = b2 + e * D_;

    // Per-lane A row pointers (rows of the two 16-row fragments).
    const _Float16* xr0 = xh + (size_t)toks[nlo]      * D_ + ak0;
    const _Float16* xr1 = xh + (size_t)toks[16 + nlo] * D_ + ak0;

    const size_t w1e = (size_t)e * (H_ / 16);   // expert chunk bases
    const size_t w2e = (size_t)e * (D_ / 16);

    // Persistent y accumulators: wave owns N-chunks q*8+wave, two row frags.
    v8f yacc[4][2];
    #pragma unroll
    for (int q = 0; q < 4; ++q) { yacc[q][0] = (v8f){}; yacc[q][1] = (v8f){}; }

    for (int pass = 0; pass < 4; ++pass) {
        if (pass) __syncthreads();           // prior GEMM2 reads of hs done

        // ---- GEMM1 slab: h[:, pass*512 .. +512) = relu(x @ w1 + b1) ----
        v8f hacc[4][2];
        #pragma unroll
        for (int q = 0; q < 4; ++q) { hacc[q][0] = (v8f){}; hacc[q][1] = (v8f){}; }

        for (int kt = 0; kt < D_ / 32; ++kt) {
            const int kb = kt * 32;
            v16h a0 = load_a_frag(xr0 + kb);
            v16h a1 = load_a_frag(xr1 + kb);
            #pragma unroll
            for (int q = 0; q < 4; ++q) {
                const int ncg = pass * 32 + q * 8 + wave;       // global H chunk
                const _Float16* bp = w1p
                    + (((w1e + ncg) * (D_ / 32) + kt) * 32 + lane) * (size_t)16;
                __builtin_prefetch(bp + 512, 0, 1);             // next k-tile
                v16h b = *(const v16h*)bp;
                hacc[q][0] = wmma_f16(a0, b, hacc[q][0]);
                hacc[q][1] = wmma_f16(a1, b, hacc[q][1]);
            }
        }
        // bias + relu + store h slab (C layout: row r+8*hihalf, col nlo)
        #pragma unroll
        for (int q = 0; q < 4; ++q) {
            const int nc  = q * 8 + wave;            // local chunk in slab
            const float b1v = b1e[pass * 512 + nc * 16 + nlo];
            #pragma unroll
            for (int rf = 0; rf < 2; ++rf) {
                #pragma unroll
                for (int r = 0; r < 8; ++r) {
                    const int m = rf * 16 + r + hihalf * 8;
                    float hv = hacc[q][rf][r] + b1v;
                    hv = hv > 0.f ? hv : 0.f;
                    hs[(size_t)m * 512 + nc * 16 + nlo] = (_Float16)hv;
                }
            }
        }
        __syncthreads();

        // ---- GEMM2 partial: yacc += h_slab @ w2[pass*512 : , :] ----
        for (int kt = 0; kt < 16; ++kt) {
            const int kb  = kt * 32;
            const int ktg = pass * 16 + kt;          // global w2 k-tile
            v16h a0 = load_a_frag(hs + (size_t)nlo        * 512 + kb + ak0);
            v16h a1 = load_a_frag(hs + (size_t)(16 + nlo) * 512 + kb + ak0);
            #pragma unroll
            for (int q = 0; q < 4; ++q) {
                const int nc2 = q * 8 + wave;        // output D chunk
                const _Float16* bp = w2p
                    + (((w2e + nc2) * (H_ / 32) + ktg) * 32 + lane) * (size_t)16;
                __builtin_prefetch(bp + 512, 0, 1);
                v16h b = *(const v16h*)bp;
                yacc[q][0] = wmma_f16(a0, b, yacc[q][0]);
                yacc[q][1] = wmma_f16(a1, b, yacc[q][1]);
            }
        }
    }

    // ---- Epilogue: out[token] += gate * (y + b2) ----
    #pragma unroll
    for (int q = 0; q < 4; ++q) {
        const int nb  = (q * 8 + wave) * 16;
        const float b2v = b2e[nb + nlo];
        #pragma unroll
        for (int rf = 0; rf < 2; ++rf) {
            #pragma unroll
            for (int r = 0; r < 8; ++r) {
                const int m = rf * 16 + r + hihalf * 8;
                const float yv = yacc[q][rf][r] + b2v;
                atomicAdd(&out[(size_t)toks[m] * D_ + nb + nlo], gts[m] * yv);
            }
        }
    }
}

// ---------------------------------------------------------------------------
// Fallback FFN (round-1 style): on-the-fly f32->f16, 16-token tiles.
// Used only when the workspace is too small for packed weights.
// ---------------------------------------------------------------------------
__global__ __launch_bounds__(256) void moe_ffn_fallback_kernel(
    const float* __restrict__ x,
    const float* __restrict__ w1,
    const float* __restrict__ b1,
    const float* __restrict__ w2,
    const float* __restrict__ b2,
    const int*   __restrict__ counts,
    const int*   __restrict__ lists,
    const float* __restrict__ gates,
    float*       __restrict__ out)
{
    __shared__ __align__(16) _Float16 xs[16 * D_];        // 16 KB
    __shared__ __align__(16) _Float16 hs[16 * (H_ / 2)];  // 32 KB
    __shared__ int   toks[16];
    __shared__ float gts[16];

    const int e    = blockIdx.x >> 10;
    const int tile = blockIdx.x & 1023;
    const int cnt  = counts[e];
    const int base = tile * 16;
    if (base >= cnt) return;

    const int tid = threadIdx.x;
    if (tid < 16) {
        int  i     = base + tid;
        bool valid = (i < cnt);
        int  j     = valid ? i : base;
        toks[tid] = lists[e * T_ + j];
        gts[tid]  = valid ? gates[e * T_ + i] : 0.f;
    }
    __syncthreads();

    const float* __restrict__ w1e = w1 + (size_t)e * D_ * H_;
    const float* __restrict__ w2e = w2 + (size_t)e * H_ * D_;
    const float* __restrict__ b1e = b1 + e * H_;
    const float* __restrict__ b2e = b2 + e * D_;

    for (int i = tid; i < 16 * D_; i += 256) {
        int m = i >> 9, k = i & (D_ - 1);
        xs[i] = (_Float16)x[(size_t)toks[m] * D_ + k];
    }
    __syncthreads();

    const int lane   = tid & 31;
    const int wave   = tid >> 5;
    const int nlo    = lane & 15;
    const int hihalf = lane >> 4;
    const int ak0    = hihalf * 16;
    const int bk0    = hihalf * 16;

    v8f yacc[4];
    #pragma unroll
    for (int q = 0; q < 4; ++q) yacc[q] = (v8f){};

    for (int half = 0; half < 2; ++half) {
        if (half) __syncthreads();
        for (int nc = wave; nc < (H_ / 2) / 16; nc += 8) {
            const int nb  = nc * 16;
            const int nbg = half * (H_ / 2) + nb;
            v8f acc = (v8f){};
            for (int kt = 0; kt < D_ / 32; ++kt) {
                const int kb = kt * 32;
                v16h a = load_a_frag(xs + (size_t)nlo * D_ + kb + ak0);
                const float* bp = w1e + (size_t)(kb + bk0) * H_ + nbg + nlo;
                v16h b;
                #pragma unroll
                for (int j = 0; j < 8; ++j) {
                    b[2 * j]     = (_Float16)bp[(size_t)(2 * j) * H_];
                    b[2 * j + 1] = (_Float16)bp[(size_t)(2 * j + 1) * H_];
                }
                acc = wmma_f16(a, b, acc);
            }
            const float b1v = b1e[nbg + nlo];
            #pragma unroll
            for (int r = 0; r < 8; ++r) {
                const int m = r + hihalf * 8;
                float hv = acc[r] + b1v;
                hv = hv > 0.f ? hv : 0.f;
                hs[(size_t)m * (H_ / 2) + nb + nlo] = (_Float16)hv;
            }
        }
        __syncthreads();
        #pragma unroll
        for (int q = 0; q < 4; ++q) {
            const int nb = (wave + q * 8) * 16;
            v8f acc = yacc[q];
            for (int kt = 0; kt < (H_ / 2) / 32; ++kt) {
                const int kb = kt * 32;
                v16h a = load_a_frag(hs + (size_t)nlo * (H_ / 2) + kb + ak0);
                const int kg = half * (H_ / 2) + kb + bk0;
                const float* bp = w2e + (size_t)kg * D_ + nb + nlo;
                v16h b;
                #pragma unroll
                for (int j = 0; j < 8; ++j) {
                    b[2 * j]     = (_Float16)bp[(size_t)(2 * j) * D_];
                    b[2 * j + 1] = (_Float16)bp[(size_t)(2 * j + 1) * D_];
                }
                acc = wmma_f16(a, b, acc);
            }
            yacc[q] = acc;
        }
    }

    #pragma unroll
    for (int q = 0; q < 4; ++q) {
        const int nb  = (wave + q * 8) * 16;
        const float b2v = b2e[nb + nlo];
        #pragma unroll
        for (int r = 0; r < 8; ++r) {
            const int m = r + hihalf * 8;
            const float yv = yacc[q][r] + b2v;
            atomicAdd(&out[(size_t)toks[m] * D_ + nb + nlo], gts[m] * yv);
        }
    }
}

// ---------------------------------------------------------------------------
extern "C" void kernel_launch(void* const* d_in, const int* in_sizes, int n_in,
                              void* d_out, int out_size, void* d_ws, size_t ws_size,
                              hipStream_t stream)
{
    const float* x   = (const float*)d_in[0];
    const float* rw  = (const float*)d_in[1];
    const float* rb  = (const float*)d_in[2];
    const float* w1  = (const float*)d_in[3];
    const float* b1  = (const float*)d_in[4];
    const float* w2  = (const float*)d_in[5];
    const float* b2  = (const float*)d_in[6];
    float* out = (float*)d_out;

    // Workspace layout (256B aligned blocks):
    //   counts[E], lists[E*T] i32, gates[E*T] f32, xh[T*D] f16,
    //   w1p[E*D*H] f16, w2p[E*H*D] f16
    const size_t off_counts = 0;
    const size_t off_lists  = 256;
    const size_t off_gates  = off_lists + sizeof(int)   * (size_t)E_ * T_;
    const size_t off_xh     = off_gates + sizeof(float) * (size_t)E_ * T_;
    const size_t off_w1p    = off_xh  + sizeof(_Float16) * (size_t)T_ * D_;
    const size_t off_w2p    = off_w1p + sizeof(_Float16) * (size_t)E_ * D_ * H_;
    const size_t need_fast  = off_w2p + sizeof(_Float16) * (size_t)E_ * H_ * D_;

    char*  ws     = (char*)d_ws;
    int*   counts = (int*)(ws + off_counts);
    int*   lists  = (int*)(ws + off_lists);
    float* gatesp = (float*)(ws + off_gates);

    hipMemsetAsync(counts, 0, 64, stream);
    hipMemsetAsync(out, 0, sizeof(float) * (size_t)out_size, stream);

    moe_router_kernel<<<T_ / 16, 256, 0, stream>>>(x, rw, rb, counts, lists, gatesp);

    if (ws_size >= need_fast) {
        _Float16* xh  = (_Float16*)(ws + off_xh);
        _Float16* w1p = (_Float16*)(ws + off_w1p);
        _Float16* w2p = (_Float16*)(ws + off_w2p);

        cvt_x_kernel<<<(T_ * D_ / 8) / 256, 256, 0, stream>>>(x, xh);
        // 16 * (H/16) * (D/32) * 32 lanes / 256 = 4096 blocks each
        pack_w_kernel<H_, D_><<<4096, 256, 0, stream>>>(w1, w1p);
        pack_w_kernel<D_, H_><<<4096, 256, 0, stream>>>(w2, w2p);

        moe_ffn_fast_kernel<<<E_ * (T_ / 32), 256, 0, stream>>>(
            xh, w1p, b1, w2p, b2, counts, lists, gatesp, out);
    } else {
        moe_ffn_fallback_kernel<<<E_ * (T_ / 16), 256, 0, stream>>>(
            x, w1, b1, w2, b2, counts, lists, gatesp, out);
    }
}
